// MixJKNet_14697378087202
// MI455X (gfx1250) — matrix operand order repaired and checked by reading, not compiled, not run on hardware
//
#include <hip/hip_runtime.h>
#include <hip/hip_bf16.h>

// ---------------- problem constants (match reference) ----------------
#define NN   100000
#define EE   1600000
#define INC  64
#define HC   64
#define OUTC 40
#define OUTP 48          // OUTC padded to a multiple of 16 for WMMA tiles
#define CAT  192         // 3*HC JK concat width
#define BETA 0.5f

typedef __attribute__((ext_vector_type(16))) __bf16 v16bf;
typedef __attribute__((ext_vector_type(8)))  __bf16 v8bf;
typedef __attribute__((ext_vector_type(8)))  float  v8f;

// =====================================================================
// Pack a row-major fp32 weight matrix W[K, Nsrc] into per-lane bf16
// B-operand fragments for v_wmma_f32_16x16x32_bf16.
// Fragment (nt,kt) covers B rows K=[kt*32,kt*32+32), cols N=[nt*16,nt*16+16).
// ISA B layout (32x16, 16-bit): lane n<16 holds col N=n, K=kt*32+{0..7,16..23};
// lane n+16 holds col N=n, K=kt*32+{8..15,24..31}. Stored contiguously so the GEMM
// loads one v16bf (32B) per lane.
// =====================================================================
__global__ void pack_b_frags(const float* __restrict__ W, __bf16* __restrict__ dst,
                             int K, int Nsrc)
{
    int frag  = blockIdx.x;          // frag = nt*numKt + kt
    int numKt = K >> 5;
    int nt    = frag / numKt;
    int kt    = frag - nt * numKt;
    int lane  = threadIdx.x;         // 0..31
    int n     = nt * 16 + (lane & 15);
    int kbase = (kt << 5) + ((lane >> 4) << 3);
    __bf16* o = dst + (((size_t)frag * 32 + lane) << 4);
#pragma unroll
    for (int e = 0; e < 16; ++e) {
        float v = (n < Nsrc) ? W[(size_t)(kbase + e + ((e >> 3) << 3)) * Nsrc + n] : 0.0f;
        o[e] = (__bf16)v;
    }
}

// =====================================================================
// Layer GEMM: Z[N,64] = A[N,64] @ W[64,64], one wave per 16-row strip.
// A operand per ISA 16-bit A layout: lane&15 = row M, lane>>4 selects the
// K-half; v16bf elements 0..7 = K=kh..kh+7, 8..15 = K=kh+16..kh+23.
// A is fp32 x (layer 1) or bf16 rows inside the [N,192] concat buffer.
// 2 WMMAs per output 16x16 tile (K=64), 4 col tiles -> 8 WMMAs per wave.
// =====================================================================
__global__ void gemm_layer(const float*  __restrict__ Af32,
                           const __bf16* __restrict__ Abf, int lda,
                           const __bf16* __restrict__ Bfr,
                           float* __restrict__ Z, int nStrips)
{
    int wv = (blockIdx.x * blockDim.x + threadIdx.x) >> 5;
    if (wv >= nStrips) return;                 // whole-wave uniform exit
    int lane = threadIdx.x & 31;
    int row  = (wv << 4) + (lane & 15);
    int kh   = (lane >> 4) << 3;               // 0 or 8

    v16bf a0, a1;
    if (Af32) {                                // fp32 source, convert in-register
        const float* p = Af32 + (size_t)row * lda;
#pragma unroll
        for (int i = 0; i < 8; ++i) {
            a0[i]     = (__bf16)p[kh + i];
            a0[8 + i] = (__bf16)p[kh + 16 + i];
            a1[i]     = (__bf16)p[32 + kh + i];
            a1[8 + i] = (__bf16)p[32 + kh + 16 + i];
        }
    } else {                                   // bf16 source (concat buffer)
        const __bf16* p = Abf + (size_t)row * lda + kh;
        v8bf l0 = *(const v8bf*)(p);
        v8bf h0 = *(const v8bf*)(p + 16);
        v8bf l1 = *(const v8bf*)(p + 32);
        v8bf h1 = *(const v8bf*)(p + 48);
        a0 = __builtin_shufflevector(l0, h0, 0,1,2,3,4,5,6,7,8,9,10,11,12,13,14,15);
        a1 = __builtin_shufflevector(l1, h1, 0,1,2,3,4,5,6,7,8,9,10,11,12,13,14,15);
    }

    int mrow = (wv << 4) + ((lane >> 4) << 3); // D layout: lane>>4 selects M half
    int col  = lane & 15;
#pragma unroll
    for (int nt = 0; nt < 4; ++nt) {
        v16bf b0 = *(const v16bf*)(Bfr + (((size_t)(nt * 2 + 0) * 32 + lane) << 4));
        v16bf b1 = *(const v16bf*)(Bfr + (((size_t)(nt * 2 + 1) * 32 + lane) << 4));
        v8f acc = {};
        acc = __builtin_amdgcn_wmma_f32_16x16x32_bf16(false, a0, false, b0,
                                                      (short)0, acc, false, false);
        acc = __builtin_amdgcn_wmma_f32_16x16x32_bf16(false, a1, false, b1,
                                                      (short)0, acc, false, false);
        float* zp = Z + (size_t)mrow * HC + nt * 16 + col;
#pragma unroll
        for (int r = 0; r < 8; ++r) zp[(size_t)r * HC] = acc[r];
    }
}

// =====================================================================
// Final GEMM: out[N,40] = hcat[N,192] @ Wlin[192,40] + blin.
// K=192 -> 6 A fragments; 3 col tiles (48 padded) -> 18 WMMAs per wave.
// =====================================================================
__global__ void gemm_final(const __bf16* __restrict__ A,
                           const __bf16* __restrict__ Bfr,
                           const float*  __restrict__ bias,
                           float* __restrict__ out, int nStrips)
{
    int wv = (blockIdx.x * blockDim.x + threadIdx.x) >> 5;
    if (wv >= nStrips) return;
    int lane = threadIdx.x & 31;
    int row  = (wv << 4) + (lane & 15);
    int kh   = (lane >> 4) << 3;

    v16bf a[6];
    const __bf16* p = A + (size_t)row * CAT;
#pragma unroll
    for (int kt = 0; kt < 6; ++kt) {
        const __bf16* q = p + kt * 32 + kh;
        v8bf lo = *(const v8bf*)(q);
        v8bf hi = *(const v8bf*)(q + 16);
        a[kt] = __builtin_shufflevector(lo, hi, 0,1,2,3,4,5,6,7,8,9,10,11,12,13,14,15);
    }

    int mrow = (wv << 4) + ((lane >> 4) << 3);
    int col0 = lane & 15;
#pragma unroll
    for (int nt = 0; nt < 3; ++nt) {
        v8f acc = {};
#pragma unroll
        for (int kt = 0; kt < 6; ++kt) {
            v16bf b = *(const v16bf*)(Bfr + (((size_t)(nt * 6 + kt) * 32 + lane) << 4));
            acc = __builtin_amdgcn_wmma_f32_16x16x32_bf16(false, a[kt], false, b,
                                                          (short)0, acc, false, false);
        }
        int col = nt * 16 + col0;
        if (col < OUTC) {                      // drop the 8 pad columns
            float bb = bias[col];
#pragma unroll
            for (int r = 0; r < 8; ++r)
                out[(size_t)(mrow + r) * OUTC + col] = acc[r] + bb;
        }
    }
}

// ---------------- graph part: fp32, memory/atomic bound ----------------
__global__ void clear_f4(float4* __restrict__ p, int n4)
{
    int t = blockIdx.x * blockDim.x + threadIdx.x;
    if (t < n4) p[t] = make_float4(0.f, 0.f, 0.f, 0.f);
}

// One thread per (edge, 4-channel chunk): coalesced float4 gather of z[src],
// scale by edge weight, 4 atomic f32 adds into agg[dst] (resolved in L2;
// working set fits the 192MB L2).
__global__ void edge_scatter(const float* __restrict__ z,
                             const long long* __restrict__ src,
                             const long long* __restrict__ dst,
                             const float* __restrict__ ew,
                             float* __restrict__ agg, int E)
{
    int t = blockIdx.x * blockDim.x + threadIdx.x;
    int e = t >> 4;
    if (e >= E) return;
    int c = (t & 15) << 2;
    long long s = src[e];
    long long d = dst[e];
    float w = ew[e];
    const float4 v = *(const float4*)(z + (size_t)s * HC + c);
    float* ap = agg + (size_t)d * HC + c;
    atomicAdd(ap + 0, v.x * w);
    atomicAdd(ap + 1, v.y * w);
    atomicAdd(ap + 2, v.z * w);
    atomicAdd(ap + 3, v.w * w);
}

// h = beta*z + (1-beta)*relu(z), z = agg + b ; write bf16 into concat buffer
__global__ void bias_mix(const float* __restrict__ agg, const float* __restrict__ b,
                         __bf16* __restrict__ hcat, int colOff, int total)
{
    int t = blockIdx.x * blockDim.x + threadIdx.x;
    if (t >= total) return;
    int node = t >> 6;
    int c    = t & 63;
    float zv = agg[t] + b[c];
    float h  = BETA * zv + (1.0f - BETA) * fmaxf(zv, 0.0f);
    hcat[(size_t)node * CAT + colOff + c] = (__bf16)h;
}

// =====================================================================
extern "C" void kernel_launch(void* const* d_in, const int* in_sizes, int n_in,
                              void* d_out, int out_size, void* d_ws, size_t ws_size,
                              hipStream_t stream)
{
    const float*     x    = (const float*)d_in[0];
    const long long* eidx = (const long long*)d_in[1];   // [2,E] int64
    const float*     ew   = (const float*)d_in[2];
    const float*     W1   = (const float*)d_in[3];
    const float*     b1   = (const float*)d_in[4];
    const float*     W2   = (const float*)d_in[5];
    const float*     b2   = (const float*)d_in[6];
    const float*     W3   = (const float*)d_in[7];
    const float*     b3   = (const float*)d_in[8];
    const float*     Wl   = (const float*)d_in[9];
    const float*     bl   = (const float*)d_in[10];
    float* out = (float*)d_out;

    const long long* src = eidx;
    const long long* dst = eidx + EE;

    // ---- workspace carve-up (256B aligned) ----
    char*  ws  = (char*)d_ws;
    size_t off = 0;
    auto carve = [&](size_t bytes) {
        size_t o = off;
        off = (off + bytes + 255) & ~(size_t)255;
        return o;
    };
    float*  z    = (float*) (ws + carve((size_t)NN * HC * 4));   // GEMM output
    float*  agg  = (float*) (ws + carve((size_t)NN * HC * 4));   // scatter target
    __bf16* hcat = (__bf16*)(ws + carve((size_t)NN * CAT * 2));  // JK concat (bf16)
    __bf16* w1f  = (__bf16*)(ws + carve((size_t)8  * 512 * 2));
    __bf16* w2f  = (__bf16*)(ws + carve((size_t)8  * 512 * 2));
    __bf16* w3f  = (__bf16*)(ws + carve((size_t)8  * 512 * 2));
    __bf16* wlf  = (__bf16*)(ws + carve((size_t)18 * 512 * 2));
    (void)ws_size; (void)in_sizes; (void)n_in; (void)out_size;

    const int nStrips   = NN / 16;                 // 6250 (exact)
    const int gemmBlk   = 256;                     // 8 waves/block
    const int gemmGrid  = (nStrips * 32 + gemmBlk - 1) / gemmBlk;
    const int clrGrid   = (NN * 16 + 255) / 256;   // N*64 floats as float4
    const int scatGrid  = ((size_t)EE * 16 + 255) / 256;
    const int mixGrid   = (NN * 64 + 255) / 256;

    // ---- pack weight fragments into WMMA B layout (tiny) ----
    pack_b_frags<<<8,  32, 0, stream>>>(W1, w1f, HC,  HC);
    pack_b_frags<<<8,  32, 0, stream>>>(W2, w2f, HC,  HC);
    pack_b_frags<<<8,  32, 0, stream>>>(W3, w3f, HC,  HC);
    pack_b_frags<<<18, 32, 0, stream>>>(Wl, wlf, CAT, OUTC);

    // ---- layer 1 ----
    gemm_layer<<<gemmGrid, gemmBlk, 0, stream>>>(x, nullptr, INC, w1f, z, nStrips);
    clear_f4<<<clrGrid, 256, 0, stream>>>((float4*)agg, NN * 16);
    edge_scatter<<<scatGrid, 256, 0, stream>>>(z, src, dst, ew, agg, EE);
    bias_mix<<<mixGrid, 256, 0, stream>>>(agg, b1, hcat, 0, NN * 64);

    // ---- layer 2 (A = h1 inside hcat, ld=192) ----
    gemm_layer<<<gemmGrid, gemmBlk, 0, stream>>>(nullptr, hcat + 0, CAT, w2f, z, nStrips);
    clear_f4<<<clrGrid, 256, 0, stream>>>((float4*)agg, NN * 16);
    edge_scatter<<<scatGrid, 256, 0, stream>>>(z, src, dst, ew, agg, EE);
    bias_mix<<<mixGrid, 256, 0, stream>>>(agg, b2, hcat, 64, NN * 64);

    // ---- layer 3 ----
    gemm_layer<<<gemmGrid, gemmBlk, 0, stream>>>(nullptr, hcat + 64, CAT, w3f, z, nStrips);
    clear_f4<<<clrGrid, 256, 0, stream>>>((float4*)agg, NN * 16);
    edge_scatter<<<scatGrid, 256, 0, stream>>>(z, src, dst, ew, agg, EE);
    bias_mix<<<mixGrid, 256, 0, stream>>>(agg, b3, hcat, 128, NN * 64);

    // ---- JK concat @ Wlin + blin ----
    gemm_final<<<gemmGrid, gemmBlk, 0, stream>>>(hcat, wlf, bl, out, nStrips);
}
